// SAB_LSTM_68917045232353
// MI455X (gfx1250) — matrix operand order repaired and checked by reading, not compiled, not run on hardware
//
#include <hip/hip_runtime.h>
#include <hip/hip_bf16.h>
#include <math.h>

// ---------------- problem constants ----------------
#define B_SZ    32
#define T_STEPS 384
#define I_DIM   64
#define H_DIM   128
#define N_CLS   10
#define L_BUF   (T_STEPS + 1)      // 385
#define KS_IH   2                  // K=64  -> 2 wmma k-steps
#define KS_HH   4                  // K=128 -> 4 wmma k-steps
#define NTILES  32                 // 512 gate cols / 16
#define TOPKV   5
#define EPSV    1e-7f

// ---------------- workspace layout (float indices) ----------------
// buf  : 32*385*128   history buffer (L2-resident)
// s2g  : 32*385       cached tanh(buf)*w_t[128:] per row
// attn : 32*384*128   attn context (for final FC)
// gxp  : 384*16*32*32 precomputed x@W_ih^T + biases, fragment layout
// pWhh : 65536 bf16   WMMA-B packed W_hh
// pWih : 32768 bf16   WMMA-B packed W_ih
#define BUF_OFF    0
#define S2G_OFF    (B_SZ * L_BUF * H_DIM)                    // 1,576,960
#define ATTN_OFF   (S2G_OFF + B_SZ * L_BUF)                  // 1,589,280
#define GXP_OFF    (ATTN_OFF + B_SZ * T_STEPS * H_DIM)       // 3,162,144
#define PWHH_OFF_F (GXP_OFF + T_STEPS * 16 * 32 * 32)        // 9,453,600
#define PWIH_OFF_F (PWHH_OFF_F + (NTILES * KS_HH * 32 * 16) / 2) // +32768 floats
// total ~9.50M floats ~= 36.3 MB

typedef __attribute__((ext_vector_type(16))) __bf16 v16bf;
typedef __attribute__((ext_vector_type(8)))  float  v8f;

__device__ __forceinline__ float sigf(float x) { return 1.0f / (1.0f + __expf(-x)); }

// ---------------------------------------------------------------------------
// Pack W_ih and W_hh (row-major, g = act @ W^T) into WMMA B-fragment layout:
// lane l holds N = nt*16 + (l%16); its 16 bf16 elems are K = ks*32 + 8*(l/16) + e + (e>=8 ? 8 : 0).
// ---------------------------------------------------------------------------
__global__ void sab_pack_weights(const float* __restrict__ Wih,
                                 const float* __restrict__ Whh,
                                 __bf16* __restrict__ pWih,
                                 __bf16* __restrict__ pWhh) {
    int t = blockIdx.x * blockDim.x + threadIdx.x;
    const int N_IH = NTILES * KS_IH * 32;   // 2048
    const int N_HH = NTILES * KS_HH * 32;   // 4096
    if (t < N_IH) {
        int lane = t & 31;
        int ks   = (t >> 5) % KS_IH;
        int nt   = t / (32 * KS_IH);
        int k    = ks * 32 + 8 * (lane >> 4);
        __bf16* dst = pWih + (size_t)t * 16;
#pragma unroll
        for (int e = 0; e < 16; ++e)
            dst[e] = (__bf16)Wih[(size_t)(nt * 16 + (lane & 15)) * I_DIM + k + e + ((e >= 8) ? 8 : 0)];
    } else if (t < N_IH + N_HH) {
        int t2   = t - N_IH;
        int lane = t2 & 31;
        int ks   = (t2 >> 5) % KS_HH;
        int nt   = t2 / (32 * KS_HH);
        int k    = ks * 32 + 8 * (lane >> 4);
        __bf16* dst = pWhh + (size_t)t2 * 16;
#pragma unroll
        for (int e = 0; e < 16; ++e)
            dst[e] = (__bf16)Whh[(size_t)(nt * 16 + (lane & 15)) * H_DIM + k + e + ((e >= 8) ? 8 : 0)];
    }
}

// Zero buf row 0 and s2g column 0 (the only state read before being written).
__global__ void sab_init_state(float* __restrict__ buf, float* __restrict__ s2g) {
    int t = blockIdx.x * blockDim.x + threadIdx.x;
    if (t < B_SZ * H_DIM) {
        int b = t / H_DIM, j = t % H_DIM;
        buf[(size_t)(b * L_BUF) * H_DIM + j] = 0.0f;
    }
    if (t < B_SZ) s2g[(size_t)t * L_BUF] = 0.0f;
}

// ---------------------------------------------------------------------------
// Parallel precompute (hoisted off the sequential critical path):
// gx[step] = x_t @ W_ih^T + b_ih + b_hh, stored in the accumulator-fragment
// layout of the main kernel: gxp[((step*16+wid)*32+lane)*32 + gate*8 + v].
// One block per timestep, 512 threads, same wave/tile mapping as main kernel.
// ---------------------------------------------------------------------------
__global__ __launch_bounds__(512, 1)
void sab_precompute_gx(const float* __restrict__ x,
                       const float* __restrict__ b_ih,
                       const float* __restrict__ b_hh,
                       const __bf16* __restrict__ pWih,
                       float* __restrict__ gxp) {
    __shared__ __align__(32) __bf16 aX[2 * KS_IH * 32 * 16];   // 2048 bf16
    const int step = blockIdx.x;
    const int tid  = threadIdx.x;
    const int lane = tid & 31;
    const int wid  = tid >> 5;
    const int mt   = wid >> 3;
    const int q    = wid & 7;

    for (int idx = tid; idx < 2 * KS_IH * 32 * 16; idx += 512) {
        int e   = idx & 15;
        int l   = (idx >> 4) & 31;
        int ks  = (idx >> 9) & 1;
        int mtt = idx >> 10;
        int m   = mtt * 16 + (l & 15);
        int kk  = ks * 32 + e + ((e >= 8) ? 8 : 0) + ((l >= 16) ? 8 : 0);
        aX[idx] = (__bf16)x[((size_t)m * T_STEPS + step) * I_DIM + kk];
    }
    __syncthreads();

    v8f acc0 = {}, acc1 = {}, acc2 = {}, acc3 = {};
#pragma unroll
    for (int ks = 0; ks < KS_IH; ++ks) {
        v16bf a  = *(const v16bf*)&aX[((mt * KS_IH + ks) * 32 + lane) * 16];
        v16bf b0 = *(const v16bf*)&pWih[(((size_t)(q)      * KS_IH + ks) * 32 + lane) * 16];
        v16bf b1 = *(const v16bf*)&pWih[(((size_t)(q + 8)  * KS_IH + ks) * 32 + lane) * 16];
        v16bf b2 = *(const v16bf*)&pWih[(((size_t)(q + 16) * KS_IH + ks) * 32 + lane) * 16];
        v16bf b3 = *(const v16bf*)&pWih[(((size_t)(q + 24) * KS_IH + ks) * 32 + lane) * 16];
        acc0 = __builtin_amdgcn_wmma_f32_16x16x32_bf16(false, a, false, b0, (short)0, acc0, false, false);
        acc1 = __builtin_amdgcn_wmma_f32_16x16x32_bf16(false, a, false, b1, (short)0, acc1, false, false);
        acc2 = __builtin_amdgcn_wmma_f32_16x16x32_bf16(false, a, false, b2, (short)0, acc2, false, false);
        acc3 = __builtin_amdgcn_wmma_f32_16x16x32_bf16(false, a, false, b3, (short)0, acc3, false, false);
    }
    const int jj = q * 16 + (lane & 15);
    const float biasI = b_ih[jj]       + b_hh[jj];
    const float biasF = b_ih[128 + jj] + b_hh[128 + jj];
    const float biasG = b_ih[256 + jj] + b_hh[256 + jj];
    const float biasO = b_ih[384 + jj] + b_hh[384 + jj];
    float* dst = gxp + (((size_t)step * 16 + wid) * 32 + lane) * 32;
#pragma unroll
    for (int v = 0; v < 8; ++v) {
        dst[v]      = acc0[v] + biasI;
        dst[8 + v]  = acc1[v] + biasF;
        dst[16 + v] = acc2[v] + biasG;
        dst[24 + v] = acc3[v] + biasO;
    }
}

// ---------------------------------------------------------------------------
// Persistent single-workgroup recurrent kernel. 512 threads = 16 waves.
// Per step only h @ W_hh^T remains (16 WMMAs); W_hh fragments (128 VGPRs) stay
// register-resident across all 384 steps; c-state never leaves VGPRs.
// ---------------------------------------------------------------------------
__global__ __launch_bounds__(512, 1)
void sab_lstm_main(const float* __restrict__ wt,
                   const float* __restrict__ gxp,
                   float* __restrict__ buf,
                   float* __restrict__ s2g,
                   float* __restrict__ attn_all,
                   const __bf16* __restrict__ pWhh,
                   float* __restrict__ hs_out) {
    __shared__ __align__(32) __bf16 aA[2 * KS_HH * 32 * 16];   // 8 KB h-fragments
    __shared__ float h_lds[B_SZ * H_DIM];                      // 16 KB
    __shared__ float part_lds[B_SZ * 16];                      // 2 KB reduction tree
    __shared__ float s1_lds[B_SZ];
    __shared__ float delta_lds[B_SZ];

    const int tid  = threadIdx.x;
    const int lane = tid & 31;
    const int wid  = tid >> 5;                 // 0..15
    const int mt   = wid >> 3;                 // batch tile 0/1
    const int q    = wid & 7;                  // column-group
    const int jj   = q * 16 + (lane & 15);     // hidden column j
    const int bb   = mt * 16 + ((lane >> 4) << 3);
    const int bA   = tid >> 4;                 // attention row (0..31)
    const int j0   = (tid & 15) * 8;           // attention column base

    float c_reg[8];
#pragma unroll
    for (int v = 0; v < 8; ++v) c_reg[v] = 0.0f;
    for (int i = tid; i < B_SZ * H_DIM; i += 512) h_lds[i] = 0.0f;
    __syncthreads();

    for (int step = 0; step < T_STEPS; ++step) {
        const int r = step + 1;

        // ---- phase 1: stage A = bf16(h) in WMMA A-fragment layout ----
        for (int idx = tid; idx < 2 * KS_HH * 32 * 16; idx += 512) {
            int e   = idx & 15;
            int l   = (idx >> 4) & 31;
            int ks  = (idx >> 9) & 3;
            int mtt = idx >> 11;
            int m   = mtt * 16 + (l & 15);
            int kk  = ks * 32 + e + ((e >= 8) ? 8 : 0) + ((l >= 16) ? 8 : 0);
            aA[idx] = (__bf16)h_lds[m * H_DIM + kk];
        }
        __syncthreads();

        // ---- phase 2: g_hh = h @ W_hh^T via bf16 WMMA (f32 acc) ----
        v8f acc0 = {}, acc1 = {}, acc2 = {}, acc3 = {};
#pragma unroll
        for (int ks = 0; ks < KS_HH; ++ks) {
            v16bf a  = *(const v16bf*)&aA[((mt * KS_HH + ks) * 32 + lane) * 16];
            v16bf b0 = *(const v16bf*)&pWhh[(((size_t)(q)      * KS_HH + ks) * 32 + lane) * 16];
            v16bf b1 = *(const v16bf*)&pWhh[(((size_t)(q + 8)  * KS_HH + ks) * 32 + lane) * 16];
            v16bf b2 = *(const v16bf*)&pWhh[(((size_t)(q + 16) * KS_HH + ks) * 32 + lane) * 16];
            v16bf b3 = *(const v16bf*)&pWhh[(((size_t)(q + 24) * KS_HH + ks) * 32 + lane) * 16];
            acc0 = __builtin_amdgcn_wmma_f32_16x16x32_bf16(false, a, false, b0, (short)0, acc0, false, false);
            acc1 = __builtin_amdgcn_wmma_f32_16x16x32_bf16(false, a, false, b1, (short)0, acc1, false, false);
            acc2 = __builtin_amdgcn_wmma_f32_16x16x32_bf16(false, a, false, b2, (short)0, acc2, false, false);
            acc3 = __builtin_amdgcn_wmma_f32_16x16x32_bf16(false, a, false, b3, (short)0, acc3, false, false);
        }

        // ---- phase 3: add precomputed gx (fragment layout) + cell update ----
        const float* gp = gxp + (((size_t)step * 16 + wid) * 32 + lane) * 32;
        if (step + 1 < T_STEPS) __builtin_prefetch(gp + 16 * 32 * 32, 0, 1);
#pragma unroll
        for (int v = 0; v < 8; ++v) {
            float gi = acc0[v] + gp[v];
            float gf = acc1[v] + gp[8 + v];
            float gg = acc2[v] + gp[16 + v];
            float go = acc3[v] + gp[24 + v];
            float cn = sigf(gf) * c_reg[v] + sigf(gi) * tanhf(gg);
            c_reg[v] = cn;
            h_lds[(bb + v) * H_DIM + jj] = sigf(go) * tanhf(cn);   // h_pre
        }
        __syncthreads();

        // ---- phase 4a: parallel partials of s1[b] = tanh(h_pre)·w_t[:128] ----
        {
            float p = 0.0f;
#pragma unroll
            for (int e = 0; e < 8; ++e)
                p += tanhf(h_lds[bA * H_DIM + j0 + e]) * wt[j0 + e];
            part_lds[bA * 16 + (tid & 15)] = p;
        }
        __syncthreads();
        // ---- phase 4b: reduce ----
        if (tid < B_SZ) {
            float s = 0.0f;
#pragma unroll
            for (int e = 0; e < 16; ++e) s += part_lds[tid * 16 + e];
            s1_lds[tid] = s;
        }
        __syncthreads();

        // ---- phase 5: top-5 threshold per batch row (r > 5 path) ----
        if (r > TOPKV) {
            for (int half = 0; half < 2; ++half) {
                int b = wid + half * 16;
                float s1b = s1_lds[b];
                const float* s2row = s2g + (size_t)b * L_BUF;
                int sel0 = -1, sel1 = -1, sel2 = -1, sel3 = -1;
                float last = 0.0f;
#pragma unroll
                for (int p = 0; p < TOPKV; ++p) {
                    float lm = -INFINITY; int li = -1;
                    for (int t2 = lane; t2 < r; t2 += 32) {
                        if (t2 == sel0 || t2 == sel1 || t2 == sel2 || t2 == sel3) continue;
                        float vv = s1b + s2row[t2];
                        if (vv > lm) { lm = vv; li = t2; }
                    }
#pragma unroll
                    for (int off = 16; off >= 1; off >>= 1) {
                        float ov = __shfl_xor(lm, off, 32);
                        int   oi = __shfl_xor(li, off, 32);
                        if (ov > lm || (ov == lm && oi >= 0 && (li < 0 || oi < li))) { lm = ov; li = oi; }
                    }
                    if      (p == 0) sel0 = li;
                    else if (p == 1) sel1 = li;
                    else if (p == 2) sel2 = li;
                    else if (p == 3) sel3 = li;
                    last = lm;
                }
                if (lane == 0) delta_lds[b] = last + EPSV;
            }
        }
        __syncthreads();

        // ---- phase 6: attention context + state/output update + s2 partial ----
        {
            const float* bufrow = buf + (size_t)bA * L_BUF * H_DIM;
            const float* s2row  = s2g + (size_t)bA * L_BUF;
            float s1b = s1_lds[bA];
            float num[8];
#pragma unroll
            for (int e = 0; e < 8; ++e) num[e] = 0.0f;
            float den;
            if (r <= TOPKV) {  // masked softmax over t' < r
                float mx = -INFINITY;
                for (int t2 = 0; t2 < r; ++t2) mx = fmaxf(mx, s1b + s2row[t2]);
                float se = 0.0f;
                for (int t2 = 0; t2 < r; ++t2) se += __expf(s1b + s2row[t2] - mx);
                for (int t2 = 0; t2 < r; ++t2) {
                    float a = __expf(s1b + s2row[t2] - mx) / se;
                    const float4* p = (const float4*)(bufrow + (size_t)t2 * H_DIM + j0);
                    float4 u0 = p[0], u1 = p[1];
                    num[0] += a * u0.x; num[1] += a * u0.y; num[2] += a * u0.z; num[3] += a * u0.w;
                    num[4] += a * u1.x; num[5] += a * u1.y; num[6] += a * u1.z; num[7] += a * u1.w;
                }
                den = 1.0f;
            } else {           // sparse ReLU(score - dmat) attention
                den = 0.0f;
                for (int t2 = 0; t2 < r; ++t2) {
                    float s  = s1b + s2row[t2];
                    float wv = s - delta_lds[(bA * r + t2) & 31];
                    if (wv > 0.0f) {
                        den += wv;
                        const float4* p = (const float4*)(bufrow + (size_t)t2 * H_DIM + j0);
                        float4 u0 = p[0], u1 = p[1];
                        num[0] += wv * u0.x; num[1] += wv * u0.y; num[2] += wv * u0.z; num[3] += wv * u0.w;
                        num[4] += wv * u1.x; num[5] += wv * u1.y; num[6] += wv * u1.z; num[7] += wv * u1.w;
                    }
                }
                den += EPSV;
            }
            float inv = 1.0f / den;
            float p2 = 0.0f;
#pragma unroll
            for (int e = 0; e < 8; ++e) {
                int j = j0 + e;
                float ac = num[e] * inv;
                float hn = h_lds[bA * H_DIM + j] + ac;
                h_lds[bA * H_DIM + j] = hn;
                buf[((size_t)bA * L_BUF + r) * H_DIM + j]           = hn;
                hs_out[((size_t)bA * T_STEPS + step) * H_DIM + j]   = hn;
                attn_all[((size_t)bA * T_STEPS + step) * H_DIM + j] = ac;
                p2 += tanhf(hn) * wt[H_DIM + j];     // fused s2-partial
            }
            part_lds[bA * 16 + (tid & 15)] = p2;
        }
        __syncthreads();

        // ---- phase 7: reduce + publish score cache for new row r ----
        if (tid < B_SZ) {
            float s = 0.0f;
#pragma unroll
            for (int e = 0; e < 16; ++e) s += part_lds[tid * 16 + e];
            s2g[(size_t)tid * L_BUF + r] = s;
        }
        __syncthreads();
    }
}

// ---------------------------------------------------------------------------
// Final projection: out[b,t,cls] = [hs|attn] @ W_fc^T + b_fc
// ---------------------------------------------------------------------------
__global__ void sab_fc(const float* __restrict__ hs, const float* __restrict__ attn,
                       const float* __restrict__ Wfc, const float* __restrict__ bfc,
                       float* __restrict__ out) {
    int idx = blockIdx.x * blockDim.x + threadIdx.x;
    if (idx >= B_SZ * T_STEPS * N_CLS) return;
    int cls = idx % N_CLS;
    int bt  = idx / N_CLS;
    const float* hr = hs   + (size_t)bt * H_DIM;
    const float* ar = attn + (size_t)bt * H_DIM;
    const float* wr = Wfc  + (size_t)cls * (2 * H_DIM);
    float s = bfc[cls];
#pragma unroll 4
    for (int k = 0; k < H_DIM; ++k) s += hr[k] * wr[k];
#pragma unroll 4
    for (int k = 0; k < H_DIM; ++k) s += ar[k] * wr[H_DIM + k];
    out[idx] = s;
}

extern "C" void kernel_launch(void* const* d_in, const int* in_sizes, int n_in,
                              void* d_out, int out_size, void* d_ws, size_t ws_size,
                              hipStream_t stream) {
    const float* x   = (const float*)d_in[0];
    const float* Wih = (const float*)d_in[1];
    const float* Whh = (const float*)d_in[2];
    const float* bih = (const float*)d_in[3];
    const float* bhh = (const float*)d_in[4];
    const float* wt  = (const float*)d_in[5];
    const float* Wfc = (const float*)d_in[6];
    const float* bfc = (const float*)d_in[7];

    float* out = (float*)d_out;                         // (32,384,10)
    float* hs  = out + B_SZ * T_STEPS * N_CLS;          // (32,384,128)

    float* ws    = (float*)d_ws;                        // needs ~36.3 MB
    float* buf   = ws + BUF_OFF;
    float* s2g   = ws + S2G_OFF;
    float* attn  = ws + ATTN_OFF;
    float* gxp   = ws + GXP_OFF;
    __bf16* pWhh = (__bf16*)(ws + PWHH_OFF_F);
    __bf16* pWih = (__bf16*)(ws + PWIH_OFF_F);

    sab_pack_weights<<<dim3(24), dim3(256), 0, stream>>>(Wih, Whh, pWih, pWhh);
    sab_init_state<<<dim3(16), dim3(256), 0, stream>>>(buf, s2g);
    sab_precompute_gx<<<dim3(T_STEPS), dim3(512), 0, stream>>>(x, bih, bhh, pWih, gxp);
    sab_lstm_main<<<dim3(1), dim3(512), 0, stream>>>(wt, gxp, buf, s2g, attn, pWhh, hs);
    sab_fc<<<dim3((B_SZ * T_STEPS * N_CLS + 255) / 256), dim3(256), 0, stream>>>(hs, attn, Wfc, bfc, out);
}
